// DeformableTransformer_17085379903975
// MI455X (gfx1250) — matrix operand order repaired and checked by reading, not compiled, not run on hardware
//
#include <hip/hip_runtime.h>
#include <hip/hip_bf16.h>
#include <stdint.h>

// ---------------- problem constants (from reference) ----------------
static constexpr int kHeads  = 8;
static constexpr int kLevels = 5;
static constexpr int kPoints = 4;
static constexpr int kC      = 256;
static constexpr int kDH     = 32;     // == wave32: one lane per channel in sampling
static constexpr int kFF     = 2048;
static constexpr int kLq     = 13343;  // 100^2+50^2+25^2+13^2+7^2
static constexpr int kB      = 2;
static constexpr int kBL     = kB * kLq;

typedef __attribute__((ext_vector_type(16))) __bf16 v16bf;
typedef __attribute__((ext_vector_type(8)))  float  v8f;

__device__ __forceinline__ uint16_t f32_to_bf16(float f) {
  uint32_t x = __float_as_uint(f);
  uint32_t r = x + 0x7FFFu + ((x >> 16) & 1u);   // round-to-nearest-even
  return (uint16_t)(r >> 16);
}

union FragBF { v16bf v; uint32_t u[8]; };

// ---------------- generic bf16 WMMA GEMM ----------------
// out[M,N](f32 and/or bf16) = A[M,K](bf16,row-major) * B[K,N](bf16,row-major) + bias, opt ReLU.
// Block: 256 threads = 8 waves; block tile 128(M) x 64(N); per-wave tile 16x64 (4 WMMAs
// per K-step reusing one A fragment). K stepped by 32.
// A tile staged with CDNA5 async copy (GLOBAL_LOAD_ASYNC_TO_LDS_B128, ASYNCcnt).
__global__ __launch_bounds__(256)
void gemm_bf16_wmma(const uint16_t* __restrict__ A, const uint16_t* __restrict__ Bm,
                    const float* __restrict__ bias,
                    float* __restrict__ outF, uint16_t* __restrict__ outB,
                    int M, int N, int K, int relu)
{
  __shared__ uint16_t ldsA[128 * 32];   // [row][k]   8 KB
  __shared__ uint16_t ldsBT[64 * 32];   // [n][k]     4 KB (transposed: K pairs contiguous)

  const int t    = threadIdx.x;
  const int wave = t >> 5;
  const int lane = t & 31;
  const int half = lane >> 4;           // lane-half selects K sub-range (ISA 7.12.2)
  const int m0   = blockIdx.x * 128;
  const int n0   = blockIdx.y * 64;

  // low 32 bits of a generic pointer into LDS == LDS byte offset (aperture rule, ISA 10.2)
  const uint32_t ldsA_base = (uint32_t)(uintptr_t)(&ldsA[0]);

  v8f acc[4] = {};

  for (int k0 = 0; k0 < K; k0 += 32) {
    // ---- stage A tile: 128 rows x 32 bf16 via async copy to LDS (2 x 16B per thread) ----
#pragma unroll
    for (int i = 0; i < 2; ++i) {
      int idx = t + i * 256;            // 0..511
      int r   = idx >> 2;               // row in tile
      int qd  = idx & 3;                // 16-byte quad within the 64B row
      int gr  = m0 + r;
      uint32_t dst = ldsA_base + (uint32_t)(r * 64 + qd * 16);
      if (gr < M) {
        uint64_t src = (uint64_t)(uintptr_t)(A + (size_t)gr * K + k0 + qd * 8);
        asm volatile("global_load_async_to_lds_b128 %0, %1, off"
                     :: "v"(dst), "v"(src) : "memory");
      } else {
        uint4 z = {0u, 0u, 0u, 0u};
        *(uint4*)(&ldsA[r * 32 + qd * 8]) = z;   // zero-fill edge rows
      }
    }
    // ---- stage B tile transposed: 32(K) x 64(N) -> ldsBT[n][k], 8 bf16 per thread ----
    {
      int k = t >> 3;                   // 0..31
      int p = t & 7;                    // 8-column group
      if (n0 + p * 8 < N) {
        uint4 d = *(const uint4*)(Bm + (size_t)(k0 + k) * N + n0 + p * 8);
        const uint32_t dd[4] = {d.x, d.y, d.z, d.w};
#pragma unroll
        for (int j = 0; j < 4; ++j) {
          ldsBT[(p * 8 + 2 * j + 0) * 32 + k] = (uint16_t)(dd[j] & 0xFFFFu);
          ldsBT[(p * 8 + 2 * j + 1) * 32 + k] = (uint16_t)(dd[j] >> 16);
        }
      } else {
#pragma unroll
        for (int j = 0; j < 8; ++j) ldsBT[(p * 8 + j) * 32 + k] = 0;
      }
    }
    // prefetch next A slab while we compute (global_prefetch_b8)
    if (k0 + 32 < K) {
      int r  = t >> 2;
      int gr = m0 + r;
      if (gr < M) __builtin_prefetch(A + (size_t)gr * K + (k0 + 32) + (t & 3) * 8, 0, 1);
    }
    asm volatile("s_wait_asynccnt 0x0" ::: "memory");   // async A-tile landed in LDS
    __syncthreads();

    // ---- build fragments per ISA 7.12.2 16-bit layouts; 4 WMMAs reuse one A frag ----
    FragBF fa;
    const uint32_t* lA = (const uint32_t*)ldsA;
    const uint32_t* lB = (const uint32_t*)ldsBT;
    const int arow = (wave << 4) + (lane & 15);
#pragma unroll
    for (int v = 0; v < 8; ++v) {
      int ka = ((v >> 2) << 4) + (half << 3) + ((v & 3) << 1);
      fa.u[v] = lA[(arow * 32 + ka) >> 1];
    }
#pragma unroll
    for (int j = 0; j < 4; ++j) {
      FragBF fb;
#pragma unroll
      for (int v = 0; v < 8; ++v) {
        int kb = ((v >> 2) << 4) + (half << 3) + ((v & 3) << 1);
        fb.u[v] = lB[(((j << 4) + (lane & 15)) * 32 + kb) >> 1];
      }
      acc[j] = __builtin_amdgcn_wmma_f32_16x16x32_bf16(
          /*neg_a=*/false, fa.v, /*neg_b=*/false, fb.v,
          /*c_mod=*/(short)0, acc[j], /*reuse_a=*/false, /*reuse_b=*/false);
    }
    __syncthreads();
  }

  // ---- epilogue: bias (+ReLU), dual-format store. C/D layout: N=lane&15, M=v+8*half ----
#pragma unroll
  for (int j = 0; j < 4; ++j) {
    int n = n0 + (j << 4) + (lane & 15);
    if (n < N) {
      float bv = bias ? bias[n] : 0.f;
#pragma unroll
      for (int v = 0; v < 8; ++v) {
        int m = m0 + (wave << 4) + v + (half << 3);
        if (m < M) {
          float r = acc[j][v] + bv;
          if (relu) r = fmaxf(r, 0.f);
          if (outF) outF[(size_t)m * N + n] = r;
          if (outB) outB[(size_t)m * N + n] = f32_to_bf16(r);
        }
      }
    }
  }
}

// ---------------- q = src + pos (bf16), src -> bf16 ----------------
__global__ __launch_bounds__(256)
void add_pos_cvt(const float* __restrict__ src, const float* __restrict__ pos,
                 uint16_t* __restrict__ qbf, uint16_t* __restrict__ sbf)
{
  size_t i = (size_t)blockIdx.x * 256 + threadIdx.x;   // grid sized exactly BL*256/256
  float s = src[i];
  qbf[i] = f32_to_bf16(s + pos[i]);
  sbf[i] = f32_to_bf16(s);
}

// ---------------- f32 -> bf16 weight conversion ----------------
__global__ __launch_bounds__(256)
void f2b(const float* __restrict__ in, uint16_t* __restrict__ out, int n)
{
  int i = blockIdx.x * 256 + threadIdx.x;
  if (i < n) out[i] = f32_to_bf16(in[i]);
}

// ---------------- softmax over LEVELS*POINTS = 20, one thread per (b,q,h) ----------------
__global__ __launch_bounds__(256)
void softmax20(float* __restrict__ logits)
{
  int tid = blockIdx.x * 256 + threadIdx.x;            // grid = BL*8/256 rounded down (BL*8 % 256 == 240)
  int bq = tid >> 3, h = tid & 7;
  float* p = logits + (size_t)bq * (kHeads * kLevels * kPoints) + h * (kLevels * kPoints);
  float v[20];
  float mx = -1e30f;
#pragma unroll
  for (int i = 0; i < 20; ++i) { v[i] = p[i]; mx = fmaxf(mx, v[i]); }
  float s = 0.f;
#pragma unroll
  for (int i = 0; i < 20; ++i) { v[i] = __expf(v[i] - mx); s += v[i]; }
  float inv = 1.f / s;
#pragma unroll
  for (int i = 0; i < 20; ++i) p[i] = v[i] * inv;
}

// ---------------- MS-deformable bilinear sampling ----------------
// One wave per (b,q,h); lane = channel d (DH==32==wave32). Each gather corner is a
// 128B coalesced read across the wave (value layout [B*L, H*DH]).
__global__ __launch_bounds__(256)
void msdeform_sample(const float* __restrict__ value, const float* __restrict__ offs,
                     const float* __restrict__ attn, const float* __restrict__ ref,
                     uint16_t* __restrict__ accb)
{
  const int HsA[5] = {100, 50, 25, 13, 7};
  const int WsA[5] = {100, 50, 25, 13, 7};
  const int LS[5]  = {0, 10000, 12500, 13125, 13294};

  int wid  = blockIdx.x * 8 + (threadIdx.x >> 5);      // grid sized exactly BL (8 waves/blk)
  int lane = threadIdx.x & 31;
  int h  = wid & 7;
  int bq = wid >> 3;                                   // 0..BL-1
  int b  = bq / kLq;

  const float* refp = ref  + (size_t)bq * kLevels * 2;
  const float* op   = offs + (size_t)bq * (kHeads * kLevels * kPoints * 2) + h * (kLevels * kPoints * 2);
  const float* ap   = attn + (size_t)bq * (kHeads * kLevels * kPoints)     + h * (kLevels * kPoints);

  float a = 0.f;
#pragma unroll
  for (int l = 0; l < kLevels; ++l) {
    const int Hs = HsA[l], Ws = WsA[l];
    const float* vb = value + ((size_t)b * kLq + LS[l]) * kC + h * kDH + lane;
    float rx = refp[l * 2 + 0], ry = refp[l * 2 + 1];
#pragma unroll
    for (int p = 0; p < kPoints; ++p) {
      // loc*W - 0.5 == ref*W + off - 0.5  (offsets are divided by (W,H) in reference)
      float px = rx * (float)Ws + op[(l * kPoints + p) * 2 + 0] - 0.5f;
      float py = ry * (float)Hs + op[(l * kPoints + p) * 2 + 1] - 0.5f;
      float w  = ap[l * kPoints + p];
      float fx0 = floorf(px), fy0 = floorf(py);
      float lx = px - fx0, ly = py - fy0;
      int x0 = (int)fx0, y0 = (int)fy0;
#pragma unroll
      for (int dy = 0; dy < 2; ++dy) {
#pragma unroll
        for (int dx = 0; dx < 2; ++dx) {
          int xi = x0 + dx, yi = y0 + dy;
          bool valid = (xi >= 0) && (xi < Ws) && (yi >= 0) && (yi < Hs);
          float cw = (dx ? lx : 1.f - lx) * (dy ? ly : 1.f - ly);
          if (valid) {
            int xc = min(max(xi, 0), Ws - 1);
            int yc = min(max(yi, 0), Hs - 1);
            a += w * cw * vb[(size_t)(yc * Ws + xc) * kC];
          }
        }
      }
    }
  }
  accb[(size_t)bq * kC + h * kDH + lane] = f32_to_bf16(a);
}

// ---------------- fused residual + LayerNorm (wave per row, wave32 shfl reduce) ----------------
__global__ __launch_bounds__(256)
void ln_res(const float* __restrict__ a, const float* __restrict__ r,
            const float* __restrict__ g, const float* __restrict__ beta,
            float* __restrict__ outF, uint16_t* __restrict__ outB)
{
  int row = blockIdx.x * 8 + (threadIdx.x >> 5);
  if (row >= kBL) return;
  int lane = threadIdx.x & 31;
  float vals[8];
  float s = 0.f, s2 = 0.f;
#pragma unroll
  for (int i = 0; i < 8; ++i) {
    int c = i * 32 + lane;
    float v = a[(size_t)row * kC + c] + r[(size_t)row * kC + c];
    vals[i] = v; s += v; s2 += v * v;
  }
#pragma unroll
  for (int m = 16; m >= 1; m >>= 1) {
    s  += __shfl_xor(s,  m, 32);
    s2 += __shfl_xor(s2, m, 32);
  }
  float mean = s * (1.f / kC);
  float var  = s2 * (1.f / kC) - mean * mean;
  float rstd = rsqrtf(var + 1e-5f);
#pragma unroll
  for (int i = 0; i < 8; ++i) {
    int c = i * 32 + lane;
    float o = (vals[i] - mean) * rstd * g[c] + beta[c];
    if (outF) outF[(size_t)row * kC + c] = o;
    if (outB) outB[(size_t)row * kC + c] = f32_to_bf16(o);
  }
}

// ---------------- host launch ----------------
static void launch_gemm(const uint16_t* A, const uint16_t* Bm, const float* bias,
                        float* oF, uint16_t* oB, int M, int N, int K, int relu,
                        hipStream_t s)
{
  dim3 g((M + 127) / 128, (N + 63) / 64);
  hipLaunchKernelGGL(gemm_bf16_wmma, g, dim3(256), 0, s, A, Bm, bias, oF, oB, M, N, K, relu);
}

extern "C" void kernel_launch(void* const* d_in, const int* in_sizes, int n_in,
                              void* d_out, int out_size, void* d_ws, size_t ws_size,
                              hipStream_t stream)
{
  // inputs in setup_inputs() order
  const float* src  = (const float*)d_in[0];
  const float* pos  = (const float*)d_in[1];
  const float* ref  = (const float*)d_in[2];
  const float* wv   = (const float*)d_in[3];
  const float* bv   = (const float*)d_in[4];
  const float* wo   = (const float*)d_in[5];
  const float* bo   = (const float*)d_in[6];
  const float* wa   = (const float*)d_in[7];
  const float* ba   = (const float*)d_in[8];
  const float* wout = (const float*)d_in[9];
  const float* bout = (const float*)d_in[10];
  const float* ln1g = (const float*)d_in[11];
  const float* ln1b = (const float*)d_in[12];
  const float* w1   = (const float*)d_in[13];
  const float* b1   = (const float*)d_in[14];
  const float* w2   = (const float*)d_in[15];
  const float* b2   = (const float*)d_in[16];
  const float* ln2g = (const float*)d_in[17];
  const float* ln2b = (const float*)d_in[18];
  float* out = (float*)d_out;

  // workspace carve-up
  char* wsb = (char*)d_ws;
  size_t off = 0;
  auto alloc = [&](size_t bytes) -> void* {
    void* p = (void*)(wsb + off);
    off += (bytes + 255) & ~(size_t)255;
    return p;
  };
  uint16_t* qbf     = (uint16_t*)alloc((size_t)kBL * kC * 2);
  uint16_t* sbf     = (uint16_t*)alloc((size_t)kBL * kC * 2);
  uint16_t* wv_b    = (uint16_t*)alloc((size_t)kC * kC * 2);
  uint16_t* wo_b    = (uint16_t*)alloc((size_t)kC * (kHeads * kLevels * kPoints * 2) * 2);
  uint16_t* wa_b    = (uint16_t*)alloc((size_t)kC * (kHeads * kLevels * kPoints) * 2);
  uint16_t* wout_b  = (uint16_t*)alloc((size_t)kC * kC * 2);
  uint16_t* w1_b    = (uint16_t*)alloc((size_t)kC * kFF * 2);
  uint16_t* w2_b    = (uint16_t*)alloc((size_t)kFF * kC * 2);
  float*    value   = (float*)   alloc((size_t)kBL * kC * 4);
  float*    offsv   = (float*)   alloc((size_t)kBL * (kHeads * kLevels * kPoints * 2) * 4);
  float*    logits  = (float*)   alloc((size_t)kBL * (kHeads * kLevels * kPoints) * 4);
  uint16_t* acc_b   = (uint16_t*)alloc((size_t)kBL * kC * 2);
  float*    attnout = (float*)   alloc((size_t)kBL * kC * 4);
  float*    x       = (float*)   alloc((size_t)kBL * kC * 4);
  uint16_t* x_b     = (uint16_t*)alloc((size_t)kBL * kC * 2);
  uint16_t* h1_b    = (uint16_t*)alloc((size_t)kBL * kFF * 2);
  float*    f2out   = (float*)   alloc((size_t)kBL * kC * 4);
  (void)ws_size; (void)n_in; (void)in_sizes; (void)out_size;

  // 1) weights -> bf16
  auto cvt = [&](const float* in, uint16_t* o, int n) {
    hipLaunchKernelGGL(f2b, dim3((n + 255) / 256), dim3(256), 0, stream, in, o, n);
  };
  cvt(wv,   wv_b,   kC * kC);
  cvt(wo,   wo_b,   kC * kHeads * kLevels * kPoints * 2);
  cvt(wa,   wa_b,   kC * kHeads * kLevels * kPoints);
  cvt(wout, wout_b, kC * kC);
  cvt(w1,   w1_b,   kC * kFF);
  cvt(w2,   w2_b,   kFF * kC);

  // 2) q = src + pos (bf16), src bf16
  hipLaunchKernelGGL(add_pos_cvt, dim3(kBL), dim3(256), 0, stream, src, pos, qbf, sbf);

  // 3) projections (WMMA)
  launch_gemm(sbf, wv_b, bv, value,  nullptr, kBL, kC,                                kC, 0, stream);
  launch_gemm(qbf, wo_b, bo, offsv,  nullptr, kBL, kHeads * kLevels * kPoints * 2,    kC, 0, stream);
  launch_gemm(qbf, wa_b, ba, logits, nullptr, kBL, kHeads * kLevels * kPoints,        kC, 0, stream);

  // 4) softmax over 20 points per (b,q,h)
  hipLaunchKernelGGL(softmax20, dim3(kBL * kHeads / 256), dim3(256), 0, stream, logits);

  // 5) deformable bilinear sampling -> acc (bf16)
  hipLaunchKernelGGL(msdeform_sample, dim3(kBL), dim3(256), 0, stream,
                     value, offsv, logits, ref, acc_b);

  // 6) output projection (WMMA)
  launch_gemm(acc_b, wout_b, bout, attnout, nullptr, kBL, kC, kC, 0, stream);

  // 7) x = LN(src + attnout), keep f32 + bf16
  hipLaunchKernelGGL(ln_res, dim3((kBL + 7) / 8), dim3(256), 0, stream,
                     src, attnout, ln1g, ln1b, x, x_b);

  // 8) FFN: h1 = relu(x @ w1 + b1) bf16 ; f2 = h1 @ w2 + b2 f32   (WMMA)
  launch_gemm(x_b,  w1_b, b1, nullptr, h1_b, kBL, kFF, kC,  1, stream);
  launch_gemm(h1_b, w2_b, b2, f2out,  nullptr, kBL, kC,  kFF, 0, stream);

  // 9) out = LN(x + f2)
  hipLaunchKernelGGL(ln_res, dim3((kBL + 7) / 8), dim3(256), 0, stream,
                     x, f2out, ln2g, ln2b, out, nullptr);
}
